// SelfAttention_14328010900191
// MI455X (gfx1250) — matrix operand order repaired and checked by reading, not compiled, not run on hardware
//
#include <hip/hip_runtime.h>
#include <hip/hip_bf16.h>
#include <math.h>

// ---------------------------------------------------------------------------
// SAGAN self-attention block, WMMA f16 pipeline for gfx1250 (MI455X, wave32)
//   B=8, H=W=64, C=512 ; N=4096 queries, M=1024 keys, Ck=64, Cv=256
// ---------------------------------------------------------------------------

typedef _Float16 hv8  __attribute__((ext_vector_type(8)));
typedef _Float16 v16h __attribute__((ext_vector_type(16)));
typedef float    v8f  __attribute__((ext_vector_type(8)));
typedef float    v4f  __attribute__((ext_vector_type(4)));

// ---- workspace layout (units: _Float16 elements). Total ~31 MB. ----
#define OFF_WFT   ((size_t)0)          //  64x512
#define OFF_WGT   ((size_t)32768)      //  64x512
#define OFF_WHT   ((size_t)65536)      // 256x512
#define OFF_WOT   ((size_t)196608)     // 512x256
#define OFF_G     ((size_t)327680)     // 32768x64
#define OFF_FULLF ((size_t)2424832)    // 32768x64   (pre-pool f)
#define OFF_FULLH ((size_t)4521984)    // 32768x256  (pre-pool h)
#define OFF_FPOOL ((size_t)12910592)   //  8192x64
#define OFF_HT    ((size_t)13434880)   // 8 x 256 x 1024 (h transposed)
#define OFF_O     OFF_FULLF            // 32768x256  (aliases dead pre-pool staging)

// ---------------------------------------------------------------------------
// Fragment loaders (wave32).
// A 16x32 f16: lanes 0-15 = M, halves {K0-7,K16-23}; lanes 16-31 = {K8-15,K24-31}
// B 32x16 f16: lanes 0-15 = N, halves {K0-7,K16-23}; lanes 16-31 = {K8-15,K24-31}
// C/D 16x16 f32: 8 VGPRs, N = lane&15, M = vgpr + 8*(lane>=16)
// ---------------------------------------------------------------------------

__device__ inline v16h load_a_f16(const _Float16* src, int ld, int m_base, int k_base) {
  int lane = threadIdx.x & 31;
  int m = lane & 15;
  int lo = k_base + ((lane >> 4) << 3);
  const _Float16* p = src + (size_t)(m_base + m) * ld + lo;
  hv8 a = *(const hv8*)p;
  hv8 b = *(const hv8*)(p + 16);
  v16h r;
#pragma unroll
  for (int i = 0; i < 8; ++i) { r[i] = a[i]; r[8 + i] = b[i]; }
  return r;
}

__device__ inline v16h load_a_f32(const float* src, int ld, int m_base, int k_base) {
  int lane = threadIdx.x & 31;
  int m = lane & 15;
  int lo = k_base + ((lane >> 4) << 3);
  const float* p = src + (size_t)(m_base + m) * ld + lo;
  v4f a0 = *(const v4f*)p;
  v4f a1 = *(const v4f*)(p + 4);
  v4f b0 = *(const v4f*)(p + 16);
  v4f b1 = *(const v4f*)(p + 20);
  v16h r;
#pragma unroll
  for (int i = 0; i < 4; ++i) {
    r[i]      = (_Float16)a0[i];
    r[4 + i]  = (_Float16)a1[i];
    r[8 + i]  = (_Float16)b0[i];
    r[12 + i] = (_Float16)b1[i];
  }
  return r;
}

// src is laid out so that B's N dimension indexes rows of src (i.e. transposed
// relative to the mathematical B); K runs contiguously along each row.
__device__ inline v16h load_b_f16(const _Float16* src, int ld, int n_base, int k_base) {
  int lane = threadIdx.x & 31;
  int n = lane & 15;
  int lo = k_base + ((lane >> 4) << 3);   // 0 or 8
  const _Float16* p = src + (size_t)(n_base + n) * ld + lo;
  hv8 a = *(const hv8*)p;
  hv8 b = *(const hv8*)(p + 16);
  v16h r;
#pragma unroll
  for (int i = 0; i < 8; ++i) { r[i] = a[i]; r[8 + i] = b[i]; }
  return r;
}

__device__ inline v8f wmma16(v16h a, v16h b, v8f c) {
  return __builtin_amdgcn_wmma_f32_16x16x32_f16(false, a, false, b, (short)0, c,
                                                false, false);
}

// ---------------------------------------------------------------------------
// Kernel: transpose-convert f32 weight [R][Cc] -> f16 [Cc][R]
// ---------------------------------------------------------------------------
__global__ void convert_w(const float* __restrict__ src, _Float16* __restrict__ dst,
                          int R, int Cc) {
  int idx = blockIdx.x * blockDim.x + threadIdx.x;
  if (idx >= R * Cc) return;
  int c = idx % Cc, r = idx / Cc;
  dst[(size_t)c * R + r] = (_Float16)src[(size_t)r * Cc + c];
}

// ---------------------------------------------------------------------------
// Kernel: y[32768][ncols] (f16) = x[32768][512] (f32) @ W  with W^T staged f16
// One wave per 16x16 output tile, K = 512 in 16 WMMA steps.
// ---------------------------------------------------------------------------
__global__ __launch_bounds__(32) void gemm_xw(const float* __restrict__ x,
                                              const _Float16* __restrict__ WT,
                                              _Float16* __restrict__ y, int ncols) {
  int rowBase = blockIdx.x * 16;
  int colBase = blockIdx.y * 16;
  v8f acc = {0.f, 0.f, 0.f, 0.f, 0.f, 0.f, 0.f, 0.f};
#pragma unroll 4
  for (int k = 0; k < 512; k += 32) {
    v16h a = load_a_f32(x, 512, rowBase, k);
    v16h b = load_b_f16(WT, 512, colBase, k);
    acc = wmma16(a, b, acc);
  }
  int lane = threadIdx.x & 31;
  int col = colBase + (lane & 15);
  int rh = (lane >> 4) << 3;
#pragma unroll
  for (int r = 0; r < 8; ++r)
    y[(size_t)(rowBase + rh + r) * ncols + col] = (_Float16)acc[r];
}

// ---------------------------------------------------------------------------
// Kernel: 2x2 maxpool on f16 [B,64,64,ncols] -> [B,32,32,ncols]
// transposed==1 writes dst[b][col][m] (for h, so attention B-frags are contiguous)
// ---------------------------------------------------------------------------
__global__ void pool_kernel(const _Float16* __restrict__ src,
                            _Float16* __restrict__ dst, int ncols, int transposed) {
  int idx = blockIdx.x * blockDim.x + threadIdx.x;
  int total = 8 * 1024 * ncols;
  if (idx >= total) return;
  int col = idx % ncols;
  int t = idx / ncols;
  int j = t & 31, i = (t >> 5) & 31, b = t >> 10;
  const _Float16* s00 = src + (size_t)((b * 64 + 2 * i) * 64 + 2 * j) * ncols + col;
  float m0 = (float)s00[0];
  m0 = fmaxf(m0, (float)s00[(size_t)ncols]);
  m0 = fmaxf(m0, (float)s00[(size_t)64 * ncols]);
  m0 = fmaxf(m0, (float)s00[(size_t)65 * ncols]);
  int mpos = (i << 5) | j;
  if (!transposed)
    dst[(size_t)(b * 1024 + mpos) * ncols + col] = (_Float16)m0;
  else
    dst[((size_t)b * ncols + col) * 1024 + mpos] = (_Float16)m0;
}

// ---------------------------------------------------------------------------
// Kernel: fused flash attention.
// One wave per (batch, 16-query tile).  s = g(16x64) @ f^T, online softmax,
// O(16x256) += P @ h, all via v_wmma_f32_16x16x32_f16.
// ---------------------------------------------------------------------------
__global__ __launch_bounds__(32) void attn_kernel(const _Float16* __restrict__ g16,
                                                  const _Float16* __restrict__ fp,
                                                  const _Float16* __restrict__ hT,
                                                  _Float16* __restrict__ o16) {
  int b = blockIdx.x >> 8;
  int qt = blockIdx.x & 255;
  int lane = threadIdx.x & 31;
  __shared__ _Float16 P[16 * 32];

  // query fragments stay resident (K = 64 -> two A frags)
  v16h gA0 = load_a_f16(g16, 64, b * 4096 + qt * 16, 0);
  v16h gA1 = load_a_f16(g16, 64, b * 4096 + qt * 16, 32);

  float m[8], l[8];
  v8f O[16];
#pragma unroll
  for (int r = 0; r < 8; ++r) { m[r] = -INFINITY; l[r] = 0.f; }
#pragma unroll
  for (int vt = 0; vt < 16; ++vt)
#pragma unroll
    for (int r = 0; r < 8; ++r) O[vt][r] = 0.f;

  const _Float16* fb = fp + (size_t)b * 1024 * 64;
  const _Float16* hb = hT + (size_t)b * 256 * 1024;

  for (int kb = 0; kb < 1024; kb += 32) {
    if (kb + 32 < 1024) {
      __builtin_prefetch(fb + (size_t)(kb + 32) * 64, 0, 3);
      __builtin_prefetch(hb + (size_t)(kb + 32), 0, 3);
    }
    v8f s0 = {0.f, 0.f, 0.f, 0.f, 0.f, 0.f, 0.f, 0.f};
    v8f s1 = {0.f, 0.f, 0.f, 0.f, 0.f, 0.f, 0.f, 0.f};
    // scores for keys [kb, kb+16) and [kb+16, kb+32): B = f^T (f row-major)
    s0 = wmma16(gA0, load_b_f16(fb, 64, kb, 0), s0);
    s0 = wmma16(gA1, load_b_f16(fb, 64, kb, 32), s0);
    s1 = wmma16(gA0, load_b_f16(fb, 64, kb + 16, 0), s1);
    s1 = wmma16(gA1, load_b_f16(fb, 64, kb + 16, 32), s1);

    float scale[8];
    int rowh = (lane >> 4) << 3;
    int coln = lane & 15;
#pragma unroll
    for (int r = 0; r < 8; ++r) {
      float tm = fmaxf(s0[r], s1[r]);
      tm = fmaxf(tm, __shfl_xor(tm, 1, 32));
      tm = fmaxf(tm, __shfl_xor(tm, 2, 32));
      tm = fmaxf(tm, __shfl_xor(tm, 4, 32));
      tm = fmaxf(tm, __shfl_xor(tm, 8, 32));
      float mn = fmaxf(m[r], tm);
      scale[r] = __expf(m[r] - mn);
      m[r] = mn;
      float p0 = __expf(s0[r] - mn);
      float p1 = __expf(s1[r] - mn);
      float rs = p0 + p1;
      rs += __shfl_xor(rs, 1, 32);
      rs += __shfl_xor(rs, 2, 32);
      rs += __shfl_xor(rs, 4, 32);
      rs += __shfl_xor(rs, 8, 32);
      l[r] = l[r] * scale[r] + rs;
      int row = rowh + r;
      P[row * 32 + coln] = (_Float16)p0;
      P[row * 32 + 16 + coln] = (_Float16)p1;
    }
    asm volatile("s_wait_dscnt 0" ::: "memory");

    // re-read P in A layout (16x32, K = 32 keys)
    v16h pf;
    {
      int ar = lane & 15;
      int lo = (lane >> 4) << 3;
      const _Float16* pp = &P[ar * 32 + lo];
      hv8 a = *(const hv8*)pp;
      hv8 bq = *(const hv8*)(pp + 16);
#pragma unroll
      for (int i = 0; i < 8; ++i) { pf[i] = a[i]; pf[8 + i] = bq[i]; }
    }

#pragma unroll
    for (int vt = 0; vt < 16; ++vt) {
#pragma unroll
      for (int r = 0; r < 8; ++r) O[vt][r] *= scale[r];
      // B = h chunk [32 keys x 16 vals]; hT[b][d][m] makes K contiguous
      v16h bh = load_b_f16(hb, 1024, vt * 16, kb);
      O[vt] = wmma16(pf, bh, O[vt]);
    }
    asm volatile("s_wait_dscnt 0" ::: "memory");  // P reads done before next write
  }

  float inv[8];
#pragma unroll
  for (int r = 0; r < 8; ++r) inv[r] = 1.0f / l[r];
  int coln = lane & 15;
  int rowh = (lane >> 4) << 3;
#pragma unroll
  for (int vt = 0; vt < 16; ++vt)
#pragma unroll
    for (int r = 0; r < 8; ++r)
      o16[(size_t)(b * 4096 + qt * 16 + rowh + r) * 256 + vt * 16 + coln] =
          (_Float16)(O[vt][r] * inv[r]);
}

// ---------------------------------------------------------------------------
// Kernel: out = gamma * (o16 @ Wo) + x   (f32 out), K = 256 in 8 WMMA steps
// ---------------------------------------------------------------------------
__global__ __launch_bounds__(32) void final_kernel(const _Float16* __restrict__ o16,
                                                   const _Float16* __restrict__ WoT,
                                                   const float* __restrict__ x,
                                                   const float* __restrict__ gamma,
                                                   float* __restrict__ out) {
  int rowBase = blockIdx.x * 16;
  int colBase = blockIdx.y * 16;
  v8f acc = {0.f, 0.f, 0.f, 0.f, 0.f, 0.f, 0.f, 0.f};
#pragma unroll
  for (int k = 0; k < 256; k += 32) {
    v16h a = load_a_f16(o16, 256, rowBase, k);
    v16h b = load_b_f16(WoT, 256, colBase, k);
    acc = wmma16(a, b, acc);
  }
  float gm = gamma[0];
  int lane = threadIdx.x & 31;
  int col = colBase + (lane & 15);
  int rh = (lane >> 4) << 3;
#pragma unroll
  for (int r = 0; r < 8; ++r) {
    size_t idx = (size_t)(rowBase + rh + r) * 512 + col;
    out[idx] = gm * acc[r] + x[idx];
  }
}

// ---------------------------------------------------------------------------
extern "C" void kernel_launch(void* const* d_in, const int* in_sizes, int n_in,
                              void* d_out, int out_size, void* d_ws, size_t ws_size,
                              hipStream_t stream) {
  const float* x     = (const float*)d_in[0];  // [8,64,64,512]
  const float* Wf    = (const float*)d_in[1];  // [512,64]
  const float* Wg    = (const float*)d_in[2];  // [512,64]
  const float* Wh    = (const float*)d_in[3];  // [512,256]
  const float* Wo    = (const float*)d_in[4];  // [256,512]
  const float* gamma = (const float*)d_in[5];  // [1]
  float* out = (float*)d_out;

  _Float16* w = (_Float16*)d_ws;  // needs ~31 MB of workspace

  // 1) stage transposed f16 weights
  convert_w<<<(512 * 64 + 255) / 256, 256, 0, stream>>>(Wf, w + OFF_WFT, 512, 64);
  convert_w<<<(512 * 64 + 255) / 256, 256, 0, stream>>>(Wg, w + OFF_WGT, 512, 64);
  convert_w<<<(512 * 256 + 255) / 256, 256, 0, stream>>>(Wh, w + OFF_WHT, 512, 256);
  convert_w<<<(256 * 512 + 255) / 256, 256, 0, stream>>>(Wo, w + OFF_WOT, 256, 512);

  // 2) projection GEMMs (WMMA): g full-res; f,h full-res pre-pool
  gemm_xw<<<dim3(2048, 4), 32, 0, stream>>>(x, w + OFF_WGT, w + OFF_G, 64);
  gemm_xw<<<dim3(2048, 4), 32, 0, stream>>>(x, w + OFF_WFT, w + OFF_FULLF, 64);
  gemm_xw<<<dim3(2048, 16), 32, 0, stream>>>(x, w + OFF_WHT, w + OFF_FULLH, 256);

  // 3) 2x2 maxpool; h written transposed per batch [256][1024]
  pool_kernel<<<(8 * 1024 * 64 + 255) / 256, 256, 0, stream>>>(
      w + OFF_FULLF, w + OFF_FPOOL, 64, 0);
  pool_kernel<<<(8 * 1024 * 256 + 255) / 256, 256, 0, stream>>>(
      w + OFF_FULLH, w + OFF_HT, 256, 1);

  // 4) fused flash attention -> o16 (aliases dead pre-pool staging)
  attn_kernel<<<8 * 256, 32, 0, stream>>>(w + OFF_G, w + OFF_FPOOL, w + OFF_HT,
                                          w + OFF_O);

  // 5) out = gamma * (o @ Wo) + x
  final_kernel<<<dim3(2048, 32), 32, 0, stream>>>(w + OFF_O, w + OFF_WOT, x, gamma,
                                                  out);
}
